// Model_5772436046445
// MI455X (gfx1250) — matrix operand order repaired and checked by reading, not compiled, not run on hardware
//
#include <hip/hip_runtime.h>
#include <hip/hip_bf16.h>
#include <math.h>

// ---------------- problem constants ----------------
#define B_   8
#define T_   512
#define E_   256
#define H_   512
#define O_   50257
#define G4H  2048            // 4*H
#define M_   4096            // B*T

// ---------------- workspace layout (bytes, 256-aligned) ----------------
#define OFF_FCW   0UL           // fc_w bf16  : 50257*512*2 = 51,463,168
#define OFF_XG    51463168UL    // xg fp32    : 4096*2048*4 = 33,554,432
#define OFF_HS    85017600UL    // hs bf16    : 4096*512*2  =  4,194,304
#define OFF_X     89211904UL    // x bf16     : 4096*256*2  =  2,097,152
#define OFF_WIH   91309056UL    // W_ih bf16  : 2048*256*2  =  1,048,576
#define OFF_BIAS  92357632UL    // b_ih+b_hh  : 2048*4
#define OFF_HBUF  92365824UL    // h ping-pong: 2*4096*4
#define OFF_CNT   92398592UL    // barrier counter

typedef __bf16        v16bf __attribute__((ext_vector_type(16)));
typedef float         v8f   __attribute__((ext_vector_type(8)));
typedef unsigned int  u32x4 __attribute__((ext_vector_type(4)));
typedef float         f32x4 __attribute__((ext_vector_type(4)));

// ================= WMMA fragment loaders (CDNA5 wave32 layouts) =========
// A (16-bit, 16x32 MxK): lane L<16 -> row L, K-runs [k0+0..7] (v0-3) and
// [k0+16..23] (v4-7); lane L>=16 -> row L-16, K-runs [k0+8..15], [k0+24..31].
__device__ __forceinline__ v16bf load_frag_a(const __bf16* __restrict__ A,
                                             int ldk, int row0, int k0) {
    int lane = threadIdx.x & 31;
    const __bf16* p = A + (size_t)(row0 + (lane & 15)) * ldk
                        + k0 + ((lane >> 4) << 3);
    union { u32x4 q[2]; v16bf v; } u;
    u.q[0] = *(const u32x4*)(p);        // 8 bf16 = 16B
    u.q[1] = *(const u32x4*)(p + 16);   // next run, +32B
    return u.v;
}

// B (16-bit, 32x16 KxN): lane L<16 -> col L, K-runs [k0+0..7], [k0+16..23];
// lane L>=16 -> col L-16, K-runs [k0+8..15], [k0+24..31].  Bm is row-major [N][K], so a
// column of the B operand is a contiguous K-run of one Bm row.
__device__ __forceinline__ v16bf load_frag_b(const __bf16* __restrict__ Bm,
                                             int ldk, int col0, int k0, int N) {
    int lane = threadIdx.x & 31;
    int n = col0 + (lane & 15);
    if (n > N - 1) n = N - 1;           // clamp (stores are guarded)
    const __bf16* p = Bm + (size_t)n * ldk + k0 + ((lane >> 4) << 3);
    union { u32x4 q[2]; v16bf v; } u;
    u.q[0] = *(const u32x4*)(p);
    u.q[1] = *(const u32x4*)(p + 16);
    return u.v;
}

// ================= GEMM: C[m,n] = sum_k A[m,k]*Bm[n,k] + bias[n] =========
// block = 256 threads (8 waves, 2x4), tile 128(M) x 256(N); each wave owns a
// 64x64 subtile = 4x4 WMMA 16x16 tiles -> 16 b128 loads : 16 WMMAs per
// 32-wide K chunk (1:1), all through v_wmma_f32_16x16x32_bf16.
__global__ __launch_bounds__(256)
void gemm_bf16_wmma(const __bf16* __restrict__ A, const __bf16* __restrict__ Bm,
                    const float* __restrict__ bias, float* __restrict__ C,
                    int N, int K) {
    int wave = threadIdx.x >> 5;
    int wm   = wave & 1;                // 2 row groups of 64
    int wn   = wave >> 1;               // 4 col groups of 64
    int rb   = blockIdx.y * 128 + wm * 64;
    int cb   = blockIdx.x * 256 + wn * 64;

    v8f acc[4][4] = {};
    for (int k0 = 0; k0 < K; k0 += 32) {
        v16bf af[4], bf[4];
#pragma unroll
        for (int s = 0; s < 4; ++s) af[s] = load_frag_a(A, K, rb + 16 * s, k0);
#pragma unroll
        for (int u = 0; u < 4; ++u) bf[u] = load_frag_b(Bm, K, cb + 16 * u, k0, N);
#pragma unroll
        for (int s = 0; s < 4; ++s)
#pragma unroll
            for (int u = 0; u < 4; ++u)
                acc[s][u] = __builtin_amdgcn_wmma_f32_16x16x32_bf16(
                    false, af[s], false, bf[u], (short)0, acc[s][u],
                    false, false);
    }

    // D layout: VGPR r -> lanes 0-15: (m=r, n=lane); lanes 16-31: (m=8+r, n=lane-16)
    int lane = threadIdx.x & 31;
#pragma unroll
    for (int u = 0; u < 4; ++u) {
        int n = cb + 16 * u + (lane & 15);
        if (n < N) {
            float bv = bias[n];
#pragma unroll
            for (int s = 0; s < 4; ++s) {
                int mr = rb + 16 * s + ((lane >> 4) << 3);
                float* cp = C + (size_t)mr * N + n;
#pragma unroll
                for (int r = 0; r < 8; ++r) {
                    *cp = acc[s][u][r] + bv;
                    cp += N;
                }
            }
        }
    }
}

// ================= prep kernels =========================================
__global__ void k_embed(const int* __restrict__ tok,
                        const float* __restrict__ emb,
                        __bf16* __restrict__ x) {
    int i = blockIdx.x * 256 + threadIdx.x;          // i < M_*E_
    int m = i >> 8, e = i & (E_ - 1);
    x[i] = (__bf16)emb[(size_t)tok[m] * E_ + e];
}

__global__ void k_cvt(const float* __restrict__ s, __bf16* __restrict__ d, int n) {
    int i = blockIdx.x * 256 + threadIdx.x;
    if (i < n) d[i] = (__bf16)s[i];
}

__global__ void k_bias(const float* __restrict__ a, const float* __restrict__ b,
                       float* __restrict__ o) {
    int i = blockIdx.x * 256 + threadIdx.x;
    if (i < G4H) o[i] = a[i] + b[i];
}

__global__ void k_init(const float* __restrict__ h0, float* __restrict__ hbuf,
                       int* __restrict__ cnt) {
    int i = blockIdx.x * 256 + threadIdx.x;          // i < B_*H_ = 4096
    hbuf[i] = h0[i];
    if (i == 0) *cnt = 0;
}

// ================= persistent LSTM scan =================================
// 16 blocks x 256 threads; thread tid owns cell (b = tid>>9, j = tid&511).
// c stays in a register; h ping-pongs through global; per-step grid barrier
// via device-scope atomic arrive + spin.
__device__ __forceinline__ float sigm(float x) { return 1.f / (1.f + __expf(-x)); }

__global__ __launch_bounds__(256)
void lstm_scan(const float* __restrict__ xg, const float* __restrict__ W_hh,
               const float* __restrict__ c0, float* __restrict__ hbuf,
               __bf16* __restrict__ hs, float* __restrict__ tail,
               int* __restrict__ cnt) {
    int tid = blockIdx.x * 256 + threadIdx.x;        // 0..4095
    int b = tid >> 9, j = tid & (H_ - 1);
    float c = c0[tid];

    const f32x4* wi = (const f32x4*)(W_hh + (size_t)(j)          * H_);
    const f32x4* wf = (const f32x4*)(W_hh + (size_t)(H_ + j)     * H_);
    const f32x4* wg = (const f32x4*)(W_hh + (size_t)(2 * H_ + j) * H_);
    const f32x4* wo = (const f32x4*)(W_hh + (size_t)(3 * H_ + j) * H_);

    __shared__ float hsh[H_];                        // this block's batch row

    for (int t = 0; t < T_; ++t) {
        const float* hprev = hbuf + (size_t)(t & 1) * (B_ * H_) + (size_t)b * H_;
        hsh[threadIdx.x]       = hprev[threadIdx.x];
        hsh[threadIdx.x + 256] = hprev[threadIdx.x + 256];
        __syncthreads();

        size_t base = (size_t)(b * T_ + t) * G4H;
        float ai = xg[base + j];
        float af = xg[base + H_ + j];
        float ag = xg[base + 2 * H_ + j];
        float ao = xg[base + 3 * H_ + j];

        const f32x4* hv = (const f32x4*)hsh;
#pragma unroll 4
        for (int k = 0; k < H_ / 4; ++k) {
            f32x4 h4 = hv[k];
            f32x4 a = wi[k], f = wf[k], g = wg[k], o = wo[k];
            ai += h4.x * a.x + h4.y * a.y + h4.z * a.z + h4.w * a.w;
            af += h4.x * f.x + h4.y * f.y + h4.z * f.z + h4.w * f.w;
            ag += h4.x * g.x + h4.y * g.y + h4.z * g.z + h4.w * g.w;
            ao += h4.x * o.x + h4.y * o.y + h4.z * o.z + h4.w * o.w;
        }

        c = sigm(af) * c + sigm(ai) * tanhf(ag);
        float h = sigm(ao) * tanhf(c);

        hbuf[(size_t)((t + 1) & 1) * (B_ * H_) + tid] = h;
        hs[(size_t)(b * T_ + t) * H_ + j] = (__bf16)h;
        if (t == T_ - 1) { tail[tid] = h; tail[B_ * H_ + tid] = c; }

        // ---- grid-wide barrier (16 blocks, all co-resident) ----
        __threadfence();
        __syncthreads();
        if (threadIdx.x == 0) {
            __hip_atomic_fetch_add(cnt, 1, __ATOMIC_RELEASE, __HIP_MEMORY_SCOPE_AGENT);
            int target = 16 * (t + 1);
            while (__hip_atomic_load(cnt, __ATOMIC_ACQUIRE, __HIP_MEMORY_SCOPE_AGENT) < target)
                __builtin_amdgcn_s_sleep(1);
        }
        __syncthreads();
        __threadfence();
    }
}

// ================= launch ===============================================
extern "C" void kernel_launch(void* const* d_in, const int* in_sizes, int n_in,
                              void* d_out, int out_size, void* d_ws, size_t ws_size,
                              hipStream_t stream) {
    const int*   tok  = (const int*)  d_in[0];
    const float* h0   = (const float*)d_in[1];
    const float* c0   = (const float*)d_in[2];
    const float* emb  = (const float*)d_in[3];
    const float* W_ih = (const float*)d_in[4];
    const float* W_hh = (const float*)d_in[5];
    const float* b_ih = (const float*)d_in[6];
    const float* b_hh = (const float*)d_in[7];
    const float* fc_w = (const float*)d_in[8];
    const float* fc_b = (const float*)d_in[9];
    float* out = (float*)d_out;

    char* ws = (char*)d_ws;
    __bf16* fcw_bf  = (__bf16*)(ws + OFF_FCW);
    float*  xg      = (float*) (ws + OFF_XG);
    __bf16* hs_bf   = (__bf16*)(ws + OFF_HS);
    __bf16* x_bf    = (__bf16*)(ws + OFF_X);
    __bf16* wih_bf  = (__bf16*)(ws + OFF_WIH);
    float*  biasg   = (float*) (ws + OFF_BIAS);
    float*  hbuf    = (float*) (ws + OFF_HBUF);
    int*    cnt     = (int*)   (ws + OFF_CNT);

    // --- prep: embedding gather + bf16 casts + gate bias + state init ---
    k_embed<<<(M_ * E_) / 256, 256, 0, stream>>>(tok, emb, x_bf);
    k_cvt  <<<(G4H * E_) / 256, 256, 0, stream>>>(W_ih, wih_bf, G4H * E_);
    k_cvt  <<<(O_ * H_ + 255) / 256, 256, 0, stream>>>(fc_w, fcw_bf, O_ * H_);
    k_bias <<<G4H / 256, 256, 0, stream>>>(b_ih, b_hh, biasg);
    k_init <<<(B_ * H_) / 256, 256, 0, stream>>>(h0, hbuf, cnt);

    // --- xg = x @ W_ih^T + (b_ih + b_hh) : M=4096, N=2048, K=256 ---
    {
        dim3 grid(G4H / 256, M_ / 128);
        gemm_bf16_wmma<<<grid, 256, 0, stream>>>(x_bf, wih_bf, biasg, xg, G4H, E_);
    }

    // --- sequential LSTM scan (persistent, 16 blocks) ---
    float* tail = out + (size_t)M_ * O_;             // h_n then c_n
    lstm_scan<<<16, 256, 0, stream>>>(xg, W_hh, c0, hbuf, hs_bf, tail, cnt);

    // --- logits = hs @ fc_w^T + fc_b : M=4096, N=50257, K=512 ---
    {
        dim3 grid((O_ + 255) / 256, M_ / 128);
        gemm_bf16_wmma<<<grid, 256, 0, stream>>>(hs_bf, fcw_bf, fc_b, out, O_, H_);
    }

    (void)in_sizes; (void)n_in; (void)out_size; (void)ws_size;
}
